// BookBookGNN_80590766342425
// MI455X (gfx1250) — compile-verified, matrix operand order checked by reading, not run on hardware
//
#include <hip/hip_runtime.h>

typedef __attribute__((ext_vector_type(16))) __bf16 v16bf;
typedef __attribute__((ext_vector_type(8)))  float  v8f;

#define N_NODES 50000
#define N_EDGES 800000
#define IN_DIM  256
#define HID     128

// K-offset pattern for the 16-bit A/B WMMA operand layout (ISA 7.12.2):
// VGPR j<4 holds K pair (2j,2j+1); VGPR j>=4 holds (16+2(j-4), +1);
// lanes 16-31 are shifted +8 in K (kb = chunk*32 + (lane>>4)*8).
__device__ __forceinline__ int koff(int j) { return (j < 4) ? (2 * j) : (16 + 2 * (j - 4)); }

__global__ __launch_bounds__(256) void zero_f32(float* __restrict__ p, int n) {
  int i = blockIdx.x * blockDim.x + threadIdx.x;
  if (i < n) p[i] = 0.0f;
}

__global__ __launch_bounds__(256) void count_in_edges(const long long* __restrict__ dst,
                                                      float* __restrict__ cnt) {
  int e = blockIdx.x * blockDim.x + threadIdx.x;
  if (e < N_EDGES) atomicAdd(&cnt[(int)dst[e]], 1.0f);
}

// One wave per edge; each lane moves 4 floats of the 128-float row (coalesced).
// agg (25.6 MB) is L2-resident; this is the atomic-throughput-bound part.
__global__ __launch_bounds__(256) void scatter_sum(const long long* __restrict__ src,
                                                   const long long* __restrict__ dst,
                                                   const float* __restrict__ h,
                                                   float* __restrict__ agg) {
  int lane = threadIdx.x & 31;
  int wave = threadIdx.x >> 5;
  int e = blockIdx.x * 8 + wave;
  if (e >= N_EDGES) return;
  int s = (int)src[e];
  int d = (int)dst[e];
  const float* hs = h + (size_t)s * HID;
  float* ad = agg + (size_t)d * HID;
#pragma unroll
  for (int j = 0; j < 4; ++j) {
    int f = lane + 32 * j;
    atomicAdd(&ad[f], hs[f]);
  }
}

// Pack f32 row-major [M][K] -> bf16 WMMA A-operand layout:
//   out[((mtile*NC + kchunk)*32 + lane)*16 + i],  NC = K/32.
__global__ __launch_bounds__(256) void pack_a(const float* __restrict__ in,
                                              __bf16* __restrict__ out, int M, int K) {
  const int NC = K / 32;
  const int tid = blockIdx.x * 256 + threadIdx.x;
  const int total = (M / 16) * NC * 32;
  if (tid >= total) return;
  const int lane = tid & 31;
  const int c    = (tid >> 5) % NC;
  const int t    = tid / (32 * NC);
  const int row  = t * 16 + (lane & 15);
  const int kb   = c * 32 + (lane >> 4) * 8;
  const float* p = in + (size_t)row * K + kb;
  v16bf a;
#pragma unroll
  for (int j = 0; j < 8; ++j) {
    int o = koff(j);
    float2 v = *(const float2*)(p + o);
    a[2 * j]     = (__bf16)v.x;
    a[2 * j + 1] = (__bf16)v.y;
  }
  *(v16bf*)(out + (size_t)tid * 16) = a;
}

// Same as pack_a but scales row r by 1/max(cnt[r],1)  (mean aggregation, fused).
__global__ __launch_bounds__(256) void pack_mean(const float* __restrict__ in,
                                                 const float* __restrict__ cnt,
                                                 __bf16* __restrict__ out, int M, int K) {
  const int NC = K / 32;
  const int tid = blockIdx.x * 256 + threadIdx.x;
  const int total = (M / 16) * NC * 32;
  if (tid >= total) return;
  const int lane = tid & 31;
  const int c    = (tid >> 5) % NC;
  const int t    = tid / (32 * NC);
  const int row  = t * 16 + (lane & 15);
  const int kb   = c * 32 + (lane >> 4) * 8;
  const float inv = 1.0f / fmaxf(cnt[row], 1.0f);
  const float* p = in + (size_t)row * K + kb;
  v16bf a;
#pragma unroll
  for (int j = 0; j < 8; ++j) {
    int o = koff(j);
    float2 v = *(const float2*)(p + o);
    a[2 * j]     = (__bf16)(v.x * inv);
    a[2 * j + 1] = (__bf16)(v.y * inv);
  }
  *(v16bf*)(out + (size_t)tid * 16) = a;
}

// Pack f32 weights [K][128] -> bf16 WMMA B-operand layout:
//   out[((ntile*NC + kchunk)*32 + lane)*16 + i]
__global__ __launch_bounds__(256) void pack_b(const float* __restrict__ W,
                                              __bf16* __restrict__ out, int K) {
  const int NC = K / 32;
  const int tid = blockIdx.x * 256 + threadIdx.x;
  const int total = (HID / 16) * NC * 32;
  if (tid >= total) return;
  const int lane = tid & 31;
  const int c    = (tid >> 5) % NC;
  const int nt   = tid / (32 * NC);
  const int n    = nt * 16 + (lane & 15);
  const int kb   = c * 32 + (lane >> 4) * 8;
  v16bf b;
#pragma unroll
  for (int j = 0; j < 8; ++j) {
    int o = koff(j);
    b[2 * j]     = (__bf16)W[(size_t)(kb + o) * HID + n];
    b[2 * j + 1] = (__bf16)W[(size_t)(kb + o + 1) * HID + n];
  }
  *(v16bf*)(out + (size_t)tid * 16) = b;
}

// h = x @ W_proj + b_proj.  Packed bf16 operands: inner loop = 2x 32B loads + 1 wmma.
__global__ __launch_bounds__(256) void proj_gemm_wmma(const __bf16* __restrict__ Ap,
                                                      const __bf16* __restrict__ Bp,
                                                      const float* __restrict__ bias,
                                                      float* __restrict__ h) {
  const int lane = threadIdx.x & 31;
  const int wave = threadIdx.x >> 5;   // N-tile
  const int l    = lane & 15;
  const int half = lane >> 4;
  const int t    = blockIdx.x;         // M-tile
  const int n    = wave * 16 + l;
  const int NC   = IN_DIM / 32;        // 8

  v8f c;
#pragma unroll
  for (int r = 0; r < 8; ++r) c[r] = bias[n];

#pragma unroll
  for (int k = 0; k < NC; ++k) {
    v16bf a = *(const v16bf*)(Ap + ((size_t)(t * NC + k) * 32 + lane) * 16);
    v16bf b = *(const v16bf*)(Bp + ((size_t)(wave * NC + k) * 32 + lane) * 16);
    c = __builtin_amdgcn_wmma_f32_16x16x32_bf16(false, a, false, b, (short)0, c, false, false);
  }

#pragma unroll
  for (int r = 0; r < 8; ++r)
    h[(size_t)(t * 16 + r + 8 * half) * HID + n] = c[r];
}

// hout = relu( mean @ Wl + bl + hin @ Wr ): two chained WMMAs per 32-wide K step.
__global__ __launch_bounds__(256) void sage_combine_wmma(const __bf16* __restrict__ Mp,
                                                         const __bf16* __restrict__ Hp,
                                                         const __bf16* __restrict__ WlP,
                                                         const __bf16* __restrict__ WrP,
                                                         const float* __restrict__ bl,
                                                         float* __restrict__ hout) {
  const int lane = threadIdx.x & 31;
  const int wave = threadIdx.x >> 5;
  const int l    = lane & 15;
  const int half = lane >> 4;
  const int t    = blockIdx.x;
  const int n    = wave * 16 + l;
  const int NC   = HID / 32;           // 4

  v8f c;
#pragma unroll
  for (int r = 0; r < 8; ++r) c[r] = bl[n];

#pragma unroll
  for (int k = 0; k < NC; ++k) {
    v16bf am = *(const v16bf*)(Mp  + ((size_t)(t * NC + k) * 32 + lane) * 16);
    v16bf bL = *(const v16bf*)(WlP + ((size_t)(wave * NC + k) * 32 + lane) * 16);
    c = __builtin_amdgcn_wmma_f32_16x16x32_bf16(false, am, false, bL, (short)0, c, false, false);
    v16bf ah = *(const v16bf*)(Hp  + ((size_t)(t * NC + k) * 32 + lane) * 16);
    v16bf bR = *(const v16bf*)(WrP + ((size_t)(wave * NC + k) * 32 + lane) * 16);
    c = __builtin_amdgcn_wmma_f32_16x16x32_bf16(false, ah, false, bR, (short)0, c, false, false);
  }

#pragma unroll
  for (int r = 0; r < 8; ++r)
    hout[(size_t)(t * 16 + r + 8 * half) * HID + n] = fmaxf(c[r], 0.0f);
}

extern "C" void kernel_launch(void* const* d_in, const int* in_sizes, int n_in,
                              void* d_out, int out_size, void* d_ws, size_t ws_size,
                              hipStream_t stream) {
  (void)in_sizes; (void)n_in; (void)out_size; (void)ws_size;
  const float* x       = (const float*)d_in[0];
  const long long* ei  = (const long long*)d_in[1];  // int64 [2, E]
  const float* Wp      = (const float*)d_in[2];
  const float* bp      = (const float*)d_in[3];
  const float* Wl0     = (const float*)d_in[4];
  const float* bl0     = (const float*)d_in[5];
  const float* Wr0     = (const float*)d_in[6];
  const float* Wl1     = (const float*)d_in[7];
  const float* bl1     = (const float*)d_in[8];
  const float* Wr1     = (const float*)d_in[9];
  const long long* src = ei;
  const long long* dst = ei + N_EDGES;

  const size_t NH = (size_t)N_NODES * HID;          // 6.4M
  const size_t NX = (size_t)N_NODES * IN_DIM;       // 12.8M

  // workspace layout (~103 MB):
  float*  agg  = (float*)d_ws;                      // 25.6 MB
  float*  h0   = agg + NH;                          // 25.6 MB
  float*  cnt  = h0 + NH;                           // 0.2 MB
  __bf16* hp   = (__bf16*)(cnt + N_NODES);          // 12.8 MB (h0-packed, then h1-packed)
  __bf16* aggp = hp + NH;                           // 12.8 MB
  __bf16* xP   = aggp + NH;                         // 25.6 MB  (dead after proj GEMM)
  float*  h1   = (float*)xP;                        // aliases xP (safe: liveness disjoint)
  __bf16* WpP  = xP + NX;                           // 64 KB
  __bf16* Wl0P = WpP  + (size_t)IN_DIM * HID;       // 32 KB each
  __bf16* Wr0P = Wl0P + (size_t)HID * HID;
  __bf16* Wl1P = Wr0P + (size_t)HID * HID;
  __bf16* Wr1P = Wl1P + (size_t)HID * HID;
  float*  out  = (float*)d_out;

  const int NHi = (int)NH;
  const int packX = (N_NODES / 16) * (IN_DIM / 32) * 32;   // 800000
  const int packH = (N_NODES / 16) * (HID / 32) * 32;      // 400000
  const int packBp = (HID / 16) * (IN_DIM / 32) * 32;      // 2048
  const int packBh = (HID / 16) * (HID / 32) * 32;         // 1024

  // one-time packs
  pack_b<<<(packBp + 255) / 256, 256, 0, stream>>>(Wp,  WpP,  IN_DIM);
  pack_b<<<(packBh + 255) / 256, 256, 0, stream>>>(Wl0, Wl0P, HID);
  pack_b<<<(packBh + 255) / 256, 256, 0, stream>>>(Wr0, Wr0P, HID);
  pack_b<<<(packBh + 255) / 256, 256, 0, stream>>>(Wl1, Wl1P, HID);
  pack_b<<<(packBh + 255) / 256, 256, 0, stream>>>(Wr1, Wr1P, HID);
  pack_a<<<(packX + 255) / 256, 256, 0, stream>>>(x, xP, N_NODES, IN_DIM);

  zero_f32<<<(N_NODES + 255) / 256, 256, 0, stream>>>(cnt, N_NODES);
  count_in_edges<<<(N_EDGES + 255) / 256, 256, 0, stream>>>(dst, cnt);

  // h0 = x @ Wp + bp
  proj_gemm_wmma<<<N_NODES / 16, 256, 0, stream>>>(xP, WpP, bp, h0);
  pack_a<<<(packH + 255) / 256, 256, 0, stream>>>(h0, hp, N_NODES, HID);

  // layer 0
  zero_f32<<<(NHi + 255) / 256, 256, 0, stream>>>(agg, NHi);
  scatter_sum<<<N_EDGES / 8, 256, 0, stream>>>(src, dst, h0, agg);
  pack_mean<<<(packH + 255) / 256, 256, 0, stream>>>(agg, cnt, aggp, N_NODES, HID);
  sage_combine_wmma<<<N_NODES / 16, 256, 0, stream>>>(aggp, hp, Wl0P, Wr0P, bl0, h1);
  pack_a<<<(packH + 255) / 256, 256, 0, stream>>>(h1, hp, N_NODES, HID);

  // layer 1
  zero_f32<<<(NHi + 255) / 256, 256, 0, stream>>>(agg, NHi);
  scatter_sum<<<N_EDGES / 8, 256, 0, stream>>>(src, dst, h1, agg);
  pack_mean<<<(packH + 255) / 256, 256, 0, stream>>>(agg, cnt, aggp, N_NODES, HID);
  sage_combine_wmma<<<N_NODES / 16, 256, 0, stream>>>(aggp, hp, Wl1P, Wr1P, bl1, out);
}
